// ResidualRGCNLayer_18459769438916
// MI455X (gfx1250) — matrix operand order repaired and checked by reading, not compiled, not run on hardware
//
#include <hip/hip_runtime.h>
#include <cmath>

// RGCN layer: scatter-mean -> [mean | x] @ [W_rel; W_root] + bias -> GELU -> +x -> LayerNorm
// Sizes fixed by the reference.
#define N_NODES 50000
#define N_EDGES 800000
#define H 256
#define R 8
#define K_TOT ((R + 1) * H)      // 2304 = stacked K dimension (8 relations + root)
#define KT (K_TOT / 32)          // 72 K-tiles of 32 (bf16 WMMA depth)
#define LDS_AROW (K_TOT + 8)     // +8 bf16 pad -> row stride 1156 dwords (bank-conflict free)

typedef __attribute__((ext_vector_type(16))) __bf16 v16bf;
typedef __attribute__((ext_vector_type(8)))  float  v8f;

union FragB16 { v16bf v; uint4 q[2]; };

// f32 -> bf16 round-to-nearest-even
__device__ __forceinline__ unsigned short f2bf(float f) {
    unsigned int u = __float_as_uint(f);
    u += 0x7fffu + ((u >> 16) & 1u);
    return (unsigned short)(u >> 16);
}

// ---------------------------------------------------------------- zero scratch
__global__ void k_zero(float* __restrict__ p, long long n4) {
    long long i = (long long)blockIdx.x * blockDim.x + threadIdx.x;
    long long stride = (long long)gridDim.x * blockDim.x;
    float4 z = make_float4(0.f, 0.f, 0.f, 0.f);
    for (; i < n4; i += stride) ((float4*)p)[i] = z;
}

// ------------------------------------------------------- scatter-sum + counts
// 64 work-slots per edge; each slot moves a float4 of the 256-float source row.
__global__ void k_scatter(const float* __restrict__ x, const int* __restrict__ ei,
                          const int* __restrict__ et, float* __restrict__ sums,
                          float* __restrict__ cnt) {
    long long w = (long long)blockIdx.x * blockDim.x + threadIdx.x;
    long long stride = (long long)gridDim.x * blockDim.x;
    const long long total = (long long)N_EDGES * 64;
    for (; w < total; w += stride) {
        int e   = (int)(w >> 6);
        int sub = (int)(w & 63);
        int src = ei[e];
        int dst = ei[N_EDGES + e];
        int rel = et[e];
        float4 v = ((const float4*)(x + (long long)src * H))[sub];
        float* s = sums + ((long long)dst * R + rel) * H + sub * 4;
        atomicAdd(s + 0, v.x);
        atomicAdd(s + 1, v.y);
        atomicAdd(s + 2, v.z);
        atomicAdd(s + 3, v.w);
        if (sub == 0) atomicAdd(cnt + (long long)dst * R + rel, 1.0f);
    }
}

// -------------------------------------------------- in-place sums -> mean
__global__ void k_mean(float* __restrict__ sums, const float* __restrict__ cnt) {
    long long i = (long long)blockIdx.x * blockDim.x + threadIdx.x;
    long long stride = (long long)gridDim.x * blockDim.x;
    const long long total = (long long)N_NODES * R * (H / 4);
    for (; i < total; i += stride) {
        long long nr = i / (H / 4);
        int q = (int)(i % (H / 4));
        float inv = 1.0f / fmaxf(cnt[nr], 1.0f);
        float4* p = (float4*)(sums + nr * H) + q;
        float4 v = *p;
        v.x *= inv; v.y *= inv; v.z *= inv; v.w *= inv;
        *p = v;
    }
}

// ---------------- pack B = [W_rel stacked; W_root] into fragment-native bf16
// Tile (kt,nt) is a contiguous 1KB chunk: lane l owns 16 consecutive K values
// (base = kt*32 + 16*(l>=16)) at column nt*16 + (l&15) -> one 32B read per frag.
__global__ void k_packB(const float* __restrict__ Wrel, const float* __restrict__ Wroot,
                        unsigned short* __restrict__ Bp) {
    int t = blockIdx.x * blockDim.x + threadIdx.x;
    const int total = KT * 16 * 32;
    if (t >= total) return;
    int lane = t & 31;
    int tile = t >> 5;               // kt*16 + nt
    int nt = tile & 15;
    int kt = tile >> 4;
    int kbase = kt * 32 + ((lane & 16) ? 16 : 0);
    int col = nt * 16 + (lane & 15);
    unsigned short vals[16];
#pragma unroll
    for (int j = 0; j < 16; ++j) {
        int k = kbase + j;           // k<2048: W_rel[r=k>>8][h=k&255]; else W_root[k-2048]
        float f = (k < R * H) ? Wrel[(long long)k * H + col]
                              : Wroot[(long long)(k - R * H) * H + col];
        vals[j] = f2bf(f);
    }
    unsigned short* dst = Bp + (long long)tile * 512 + lane * 16;
    ((uint4*)dst)[0] = ((const uint4*)vals)[0];
    ((uint4*)dst)[1] = ((const uint4*)vals)[1];
}

// ------------- fused GEMM (bf16 WMMA, f32 acc) + bias + GELU + residual + LN
// One block = 16 rows x 256 cols. 8 waves x 2 col-tiles. A slab staged in LDS.
__global__ __launch_bounds__(256)
void k_gemm(const float* __restrict__ mean, const unsigned short* __restrict__ Bp,
            const float* __restrict__ x, const float* __restrict__ bias,
            const float* __restrict__ gamma, const float* __restrict__ beta,
            float* __restrict__ out) {
    __shared__ __align__(16) unsigned short sA[16 * LDS_AROW];   // 73,984 B
    const int m0  = blockIdx.x * 16;
    const int tid = threadIdx.x;

    // Stage A[16][2304] (mean f32 / node_states f32) -> bf16 LDS, coalesced float4 reads.
    for (int g = tid; g < 16 * (K_TOT / 4); g += 256) {
        int row = g / (K_TOT / 4);
        int k   = (g % (K_TOT / 4)) * 4;
        const float* src = (k < R * H)
            ? (mean + (long long)(m0 + row) * (R * H) + k)
            : (x    + (long long)(m0 + row) * H + (k - R * H));
        float4 v = *(const float4*)src;
        unsigned short tmp[4] = { f2bf(v.x), f2bf(v.y), f2bf(v.z), f2bf(v.w) };
        *(uint2*)(sA + row * LDS_AROW + k) = *(const uint2*)tmp;
    }
    __syncthreads();

    const int wave = tid >> 5;
    const int lane = tid & 31;
    const int rowA = lane & 15;
    const int hiA  = (lane >> 4) << 4;        // lanes 16-31 hold K offsets +16..+31
    const int nt0 = wave * 2, nt1 = wave * 2 + 1;

    v8f c0 = {}; v8f c1 = {};
    for (int kt = 0; kt < KT; ++kt) {
        FragB16 a;
        const unsigned short* ap = sA + rowA * LDS_AROW + kt * 32 + hiA;
        a.q[0] = *(const uint4*)(ap);         // K = base+0..7
        a.q[1] = *(const uint4*)(ap + 8);     // K = base+8..15
        FragB16 b0, b1;
        const unsigned short* bp0 = Bp + (long long)(kt * 16 + nt0) * 512 + lane * 16;
        const unsigned short* bp1 = Bp + (long long)(kt * 16 + nt1) * 512 + lane * 16;
        b0.q[0] = ((const uint4*)bp0)[0]; b0.q[1] = ((const uint4*)bp0)[1];
        b1.q[0] = ((const uint4*)bp1)[0]; b1.q[1] = ((const uint4*)bp1)[1];
        c0 = __builtin_amdgcn_wmma_f32_16x16x32_bf16(false, a.v, false, b0.v,
                                                     (short)0, c0, false, false);
        c1 = __builtin_amdgcn_wmma_f32_16x16x32_bf16(false, a.v, false, b1.v,
                                                     (short)0, c1, false, false);
    }
    __syncthreads();                          // done reading A slab

    // Spill conv (+bias) to LDS, re-using the A slab: xb[16][264] f32 = 16.9 KB.
    float* xb = (float*)sA;
    const int XROW = H + 8;
    {
        int col0 = nt0 * 16 + (lane & 15);
        int col1 = nt1 * 16 + (lane & 15);
        int rbase = (lane >> 4) << 3;         // VGPR i -> row i (lanes<16) / 8+i (lanes>=16)
        float bb0 = bias[col0], bb1 = bias[col1];
#pragma unroll
        for (int i = 0; i < 8; ++i) {
            xb[(rbase + i) * XROW + col0] = c0[i] + bb0;
            xb[(rbase + i) * XROW + col1] = c1[i] + bb1;
        }
    }
    __syncthreads();

    // Epilogue: each wave LayerNorms 2 rows (32 lanes x 8 cols), wave32 shfl reduce.
    for (int rr = 0; rr < 2; ++rr) {
        int row = wave * 2 + rr;
        long long grow = (long long)(m0 + row) * H;
        int cbase = lane * 8;
        float4 xa = ((const float4*)(x + grow + cbase))[0];
        float4 xc = ((const float4*)(x + grow + cbase))[1];
        float resid[8] = { xa.x, xa.y, xa.z, xa.w, xc.x, xc.y, xc.z, xc.w };
        float v[8];
        float s = 0.f, s2 = 0.f;
#pragma unroll
        for (int j = 0; j < 8; ++j) {
            float conv = xb[row * XROW + cbase + j];
            float u = 0.5f * conv * (1.0f + erff(conv * 0.70710678118654752440f));
            float xv = u + resid[j];
            v[j] = xv;
            s += xv; s2 += xv * xv;
        }
#pragma unroll
        for (int off = 16; off >= 1; off >>= 1) {
            s  += __shfl_xor(s,  off, 32);
            s2 += __shfl_xor(s2, off, 32);
        }
        float mu  = s * (1.0f / H);
        float var = s2 * (1.0f / H) - mu * mu;
        float inv = rsqrtf(var + 1e-5f);
        float o[8];
#pragma unroll
        for (int j = 0; j < 8; ++j)
            o[j] = gamma[cbase + j] * ((v[j] - mu) * inv) + beta[cbase + j];
        ((float4*)(out + grow + cbase))[0] = *(const float4*)&o[0];
        ((float4*)(out + grow + cbase))[1] = *(const float4*)&o[4];
    }
}

extern "C" void kernel_launch(void* const* d_in, const int* in_sizes, int n_in,
                              void* d_out, int out_size, void* d_ws, size_t ws_size,
                              hipStream_t stream) {
    const float* x     = (const float*)d_in[0];
    const int*   ei    = (const int*)d_in[1];
    const int*   et    = (const int*)d_in[2];
    const float* Wrel  = (const float*)d_in[3];
    const float* Wroot = (const float*)d_in[4];
    const float* bias  = (const float*)d_in[5];
    const float* gamma = (const float*)d_in[6];
    const float* beta  = (const float*)d_in[7];
    float* out = (float*)d_out;

    // Workspace: sums/mean [N*R*H] f32 | counts [N*R] f32 | packed B bf16 [2304*256]
    float* sums = (float*)d_ws;
    float* cnt  = sums + (long long)N_NODES * R * H;
    unsigned short* Bp = (unsigned short*)(cnt + (long long)N_NODES * R);

    long long zero_n4 = ((long long)N_NODES * R * H + (long long)N_NODES * R) / 4;
    k_zero<<<(unsigned)((zero_n4 + 255) / 256), 256, 0, stream>>>(sums, zero_n4);

    long long scat = (long long)N_EDGES * 64;
    k_scatter<<<(unsigned)((scat + 255) / 256), 256, 0, stream>>>(x, ei, et, sums, cnt);

    long long mn = (long long)N_NODES * R * (H / 4);
    k_mean<<<(unsigned)((mn + 255) / 256), 256, 0, stream>>>(sums, cnt);

    k_packB<<<(KT * 16 * 32 + 255) / 256, 256, 0, stream>>>(Wrel, Wroot, Bp);

    k_gemm<<<N_NODES / 16, 256, 0, stream>>>(sums, Bp, x, bias, gamma, beta, out);
}
